// LSTM_71287867179074
// MI455X (gfx1250) — matrix operand order repaired and checked by reading, not compiled, not run on hardware
//
#include <hip/hip_runtime.h>
#include <math.h>

// ---------------------------------------------------------------------------
// Types for CDNA5 WMMA (wave32): v_wmma_f32_16x16x32_bf16
// ---------------------------------------------------------------------------
typedef __attribute__((ext_vector_type(16))) __bf16 v16bf;
typedef __attribute__((ext_vector_type(8)))  __bf16 v8bf;
typedef __attribute__((ext_vector_type(8)))  float  v8f;

#define EMB   256
#define HID   512
#define SEQ   512
#define BATCH 64
#define G4    2048   // 4*HID gate columns
#define NBLK  16     // scan grid (one WGP per block; 2048/128 col slices)
#define NTHR  256    // 8 waves

// ---------------------------------------------------------------------------
// Device-wide sense-reversal barrier (16 co-resident blocks, s_sleep spin)
// ---------------------------------------------------------------------------
__device__ __forceinline__ void grid_barrier(volatile unsigned* cnt,
                                             volatile unsigned* gen,
                                             unsigned nb) {
  __syncthreads();
  if (threadIdx.x == 0) {
    __threadfence();
    unsigned g = *gen;
    if (atomicAdd((unsigned*)cnt, 1u) == nb - 1u) {
      *cnt = 0u;
      __threadfence();
      atomicAdd((unsigned*)gen, 1u);
    } else {
      while (*gen == g) { __builtin_amdgcn_s_sleep(1); }
    }
    __threadfence();
  }
  __syncthreads();
}

// ---------------------------------------------------------------------------
__global__ void k_init_meta(unsigned* meta) {
  if (threadIdx.x == 0) { meta[0] = 0u; meta[1] = 0u; }
}

// ---------------------------------------------------------------------------
// Concatenate [Wih | Whh] row-major [G4][K] and convert fp32 -> bf16
// ---------------------------------------------------------------------------
__global__ void k_prep_w(const float* __restrict__ wih,
                         const float* __restrict__ whh,
                         int in_dim, __bf16* __restrict__ dst, int K) {
  size_t total = (size_t)G4 * (size_t)K;
  for (size_t i = (size_t)blockIdx.x * blockDim.x + threadIdx.x; i < total;
       i += (size_t)gridDim.x * blockDim.x) {
    int row = (int)(i / K);
    int k   = (int)(i % K);
    float v = (k < in_dim) ? wih[(size_t)row * in_dim + k]
                           : whh[(size_t)row * HID + (k - in_dim)];
    dst[i] = (__bf16)v;
  }
}

// ---------------------------------------------------------------------------
// Embedding gather (padding_idx=0) -> Xb [S][B][E] bf16
// ---------------------------------------------------------------------------
__global__ void k_embed(const int* __restrict__ ids,
                        const float* __restrict__ emb,
                        __bf16* __restrict__ xb) {
  const int E4 = EMB / 4;
  size_t total = (size_t)SEQ * BATCH * E4;
  for (size_t i = (size_t)blockIdx.x * blockDim.x + threadIdx.x; i < total;
       i += (size_t)gridDim.x * blockDim.x) {
    int e4 = (int)(i % E4);
    size_t r = i / E4;
    int m = (int)(r % BATCH);
    int t = (int)(r / BATCH);
    int id = ids[(size_t)m * SEQ + t];
    float4 v = make_float4(0.f, 0.f, 0.f, 0.f);
    if (id != 0) v = ((const float4*)(emb + (size_t)id * EMB))[e4];
    __bf16* o = xb + (((size_t)t * BATCH + m) * EMB + (size_t)e4 * 4);
    o[0] = (__bf16)v.x; o[1] = (__bf16)v.y; o[2] = (__bf16)v.z; o[3] = (__bf16)v.w;
  }
}

// ---------------------------------------------------------------------------
// Persistent fused LSTM scan.
//   g[64,2048] = [x_t | h] @ Wcat^T + b   (bf16 WMMA, f32 accum)
// Block b owns gate columns { gt*512 + b*32 .. +32 } for gt=0..3, so the
// c/h update for hidden slice [b*32, b*32+32) is block-local (c lives in LDS).
// Weight slice (128 x K bf16) is persistent in LDS for all 512 steps.
// K-loop fully unrolled; fragments are TRIPLE-buffered (prefetch distance 2,
// 16 global_load_b128 in flight) and a sched_barrier pins the prefetch loads
// above each WMMA group so the scheduler cannot re-serialize them.
// ---------------------------------------------------------------------------
template <int IN>
__global__ __attribute__((amdgpu_flat_work_group_size(NTHR, NTHR),
                          amdgpu_waves_per_eu(2, 2)))
void k_lstm_scan(const __bf16* __restrict__ xseq,   // [S][B][IN]
                 const __bf16* __restrict__ wcat,   // [G4][K]
                 const float*  __restrict__ bias,   // [G4]
                 __bf16* h0buf, __bf16* h1buf,      // [B][H] ping-pong
                 __bf16* hs_out,                    // [S][B][H] or null
                 float*  h_last,                    // [B][H] or null
                 unsigned* meta) {
  constexpr int K = IN + HID;
  extern __shared__ char smem[];
  __bf16* Wl  = (__bf16*)smem;                              // 128*K bf16
  float* gact = (float*)(smem + (size_t)128 * K * 2);       // [4][64][32]
  float* cst  = gact + 4 * BATCH * 32;                      // [64][32]

  const int tid    = threadIdx.x;
  const int blk    = blockIdx.x;
  const int wave   = tid >> 5;
  const int lane   = tid & 31;
  const int laneM  = lane & 15;
  const int hiHalf = lane >> 4;          // 0/1
  const int gt     = wave >> 1;          // gate 0..3 (i,f,g,o)
  const int nhalf  = wave & 1;           // which 16-col half of the 32 strip
  const int jl     = nhalf * 16 + laneM; // col 0..31 within block's slice
  const int nlb    = gt * 32 + nhalf * 16;

  // ---- preload persistent weight slice into LDS (vectorized 16B) ----
  for (int i = tid; i < (128 * K) / 8; i += NTHR) {
    int nl = (i * 8) / K;
    int k  = (i * 8) % K;
    int col = (nl >> 5) * HID + blk * 32 + (nl & 31);   // global gate column
    *(v8bf*)(Wl + (size_t)nl * K + k) =
        *(const v8bf*)(wcat + (size_t)col * K + k);
  }
  // ---- zero cell state (LDS-resident for whole sequence) ----
  for (int i = tid; i < BATCH * 32; i += NTHR) cst[i] = 0.0f;
  // ---- zero h ping buffer (t=0 reads it) ----
  {
    const int per = (BATCH * HID) / NBLK;
    for (int i = tid; i < per; i += NTHR) h0buf[(size_t)blk * per + i] = (__bf16)0.0f;
  }
  grid_barrier(meta, meta + 1, NBLK);

  const float bv = bias[gt * HID + blk * 32 + jl];
  const int bk = (lane < 16) ? 0 : 16;   // B-frag K offset for this half-wave
  const int ak = (lane < 16) ? 0 : 16;   // A-frag K offset for this half-wave
  const __bf16* wrow = Wl + (size_t)(nlb + laneM) * K;

  // Per-lane 32-bit A-fragment offsets against uniform base pointers
  // (SADDR + 32-bit voffset addressing; kc folds into immediate offsets)
  int aoffX[4], aoffH[4];
  #pragma unroll
  for (int mt = 0; mt < 4; ++mt) {
    int m = mt * 16 + laneM;
    aoffX[mt] = m * IN + ak;
    aoffH[mt] = m * HID + ak;
  }

  for (int t = 0; t < SEQ; ++t) {
    const __bf16* hcur = (t & 1) ? h1buf : h0buf;
    __bf16*       hnxt = (t & 1) ? h0buf : h1buf;
    const __bf16* xt   = xseq + (size_t)t * BATCH * IN;

    // prefetch part of next timestep's activations into caches
    if (t + 1 < SEQ) {
      const char* nx = (const char*)(xseq + (size_t)(t + 1) * BATCH * IN);
      __builtin_prefetch(nx + (size_t)tid * 64, 0, 1);
    }

    v8f acc[4] = {};            // 4 M-tiles x this wave's 16-col N-tile
    v8bf alo[3][4], ahi[3][4];  // triple-buffered A fragments
    v16bf bfr[3];               // triple-buffered B fragments

    // fragment loader: kc is a compile-time constant after full unroll
    auto load_frags = [&](int kc, int buf) {
      bfr[buf] = *(const v16bf*)(wrow + kc + bk);
      const __bf16* abase = (kc < IN) ? xt : hcur;
      #pragma unroll
      for (int mt = 0; mt < 4; ++mt) {
        int off = (kc < IN) ? (aoffX[mt] + kc) : (aoffH[mt] + (kc - IN));
        alo[buf][mt] = *(const v8bf*)(abase + off);
        ahi[buf][mt] = *(const v8bf*)(abase + off + 8);
      }
    };

    load_frags(0, 0);
    load_frags(32, 1);
    #pragma unroll
    for (int kc = 0; kc < K; kc += 32) {
      const int cur = (kc >> 5) % 3;
      if (kc + 64 < K) load_frags(kc + 64, (cur + 2) % 3);
#if __has_builtin(__builtin_amdgcn_sched_barrier)
      __builtin_amdgcn_sched_barrier(0);  // keep prefetch loads above WMMAs
#endif
      #pragma unroll
      for (int mt = 0; mt < 4; ++mt) {
        v16bf afrag = __builtin_shufflevector(
            alo[cur][mt], ahi[cur][mt], 0, 1, 2, 3, 4, 5, 6, 7,
            8, 9, 10, 11, 12, 13, 14, 15);
        acc[mt] = __builtin_amdgcn_wmma_f32_16x16x32_bf16(
            false, afrag, false, bfr[cur], (short)0, acc[mt], false, false);
      }
    }

    // ---- spill gate tiles (+bias) into LDS for the cross-gate update ----
    #pragma unroll
    for (int mt = 0; mt < 4; ++mt) {
      #pragma unroll
      for (int r = 0; r < 8; ++r) {
        int m = mt * 16 + hiHalf * 8 + r;       // C layout: row r (+8 hi half)
        gact[(gt * BATCH + m) * 32 + jl] = acc[mt][r] + bv;
      }
    }
    __syncthreads();

    // ---- block-local LSTM cell update for hidden slice [blk*32, +32) ----
    for (int i = tid; i < BATCH * 32; i += NTHR) {
      int m = i >> 5, j = i & 31;
      float gi = gact[(0 * BATCH + m) * 32 + j];
      float gf = gact[(1 * BATCH + m) * 32 + j];
      float gg = gact[(2 * BATCH + m) * 32 + j];
      float go = gact[(3 * BATCH + m) * 32 + j];
      float si = 1.0f / (1.0f + __expf(-gi));
      float sf = 1.0f / (1.0f + __expf(-gf));
      float so = 1.0f / (1.0f + __expf(-go));
      float c = sf * cst[i] + si * tanhf(gg);
      float h = so * tanhf(c);
      cst[i] = c;
      int jg = blk * 32 + j;
      hnxt[(size_t)m * HID + jg] = (__bf16)h;
      if (hs_out) hs_out[((size_t)t * BATCH + m) * HID + jg] = (__bf16)h;
      if (h_last && t == SEQ - 1) h_last[(size_t)m * HID + jg] = h;
    }
    grid_barrier(meta, meta + 1, NBLK);   // publish h for next step
  }
}

// ---------------------------------------------------------------------------
// Tiny classifier head: out[64,2] = h_last @ fc_w^T + fc_b (fp32)
// ---------------------------------------------------------------------------
__global__ void k_fc(const float* __restrict__ hl, const float* __restrict__ w,
                     const float* __restrict__ b, float* __restrict__ out) {
  int tid = threadIdx.x;
  if (tid < BATCH * 2) {
    int m = tid >> 1, c = tid & 1;
    float s = b[c];
    const float* hr = hl + (size_t)m * HID;
    const float* wr = w + (size_t)c * HID;
    for (int k = 0; k < HID; ++k) s += hr[k] * wr[k];
    out[m * 2 + c] = s;
  }
}

// ---------------------------------------------------------------------------
extern "C" void kernel_launch(void* const* d_in, const int* in_sizes, int n_in,
                              void* d_out, int out_size, void* d_ws,
                              size_t ws_size, hipStream_t stream) {
  const int*   ids  = (const int*)d_in[0];
  const float* emb  = (const float*)d_in[1];
  const float* Wih0 = (const float*)d_in[2];
  const float* Whh0 = (const float*)d_in[3];
  const float* b0   = (const float*)d_in[4];
  const float* Wih1 = (const float*)d_in[5];
  const float* Whh1 = (const float*)d_in[6];
  const float* b1   = (const float*)d_in[7];
  const float* fcw  = (const float*)d_in[8];
  const float* fcb  = (const float*)d_in[9];

  // workspace carve (all chunk sizes multiples of 256B)
  char* p = (char*)d_ws;
  unsigned* meta = (unsigned*)p;                 p += 256;
  __bf16* h0   = (__bf16*)p;                     p += (size_t)BATCH * HID * 2;
  __bf16* h1   = (__bf16*)p;                     p += (size_t)BATCH * HID * 2;
  float*  hlast = (float*)p;                     p += (size_t)BATCH * HID * 4;
  __bf16* Wc0  = (__bf16*)p;                     p += (size_t)G4 * (EMB + HID) * 2;
  __bf16* Wc1  = (__bf16*)p;                     p += (size_t)G4 * (HID + HID) * 2;
  __bf16* Xb   = (__bf16*)p;                     p += (size_t)SEQ * BATCH * EMB * 2;
  __bf16* Hs0  = (__bf16*)p;                     p += (size_t)SEQ * BATCH * HID * 2;

  k_init_meta<<<1, 32, 0, stream>>>(meta);
  k_prep_w<<<1024, 256, 0, stream>>>(Wih0, Whh0, EMB, Wc0, EMB + HID);
  k_prep_w<<<1024, 256, 0, stream>>>(Wih1, Whh1, HID, Wc1, HID + HID);
  k_embed<<<2048, 256, 0, stream>>>(ids, emb, Xb);

  const size_t actBytes = (size_t)(4 * BATCH * 32 + BATCH * 32) * sizeof(float); // 40KB
  const size_t smem0 = (size_t)128 * (EMB + HID) * 2 + actBytes;  // ~232KB
  const size_t smem1 = (size_t)128 * (HID + HID) * 2 + actBytes;  // ~296KB

  k_lstm_scan<EMB><<<NBLK, NTHR, smem0, stream>>>(Xb, Wc0, b0, h0, h1,
                                                  Hs0, nullptr, meta);
  k_lstm_scan<HID><<<NBLK, NTHR, smem1, stream>>>(Hs0, Wc1, b1, h0, h1,
                                                  nullptr, hlast, meta);
  k_fc<<<1, 128, 0, stream>>>(hlast, fcw, fcb, (float*)d_out);
}
